// TernaryGRU_17892833755670
// MI455X (gfx1250) — matrix operand order repaired and checked by reading, not compiled, not run on hardware
//
#include <hip/hip_runtime.h>
#include <hip/hip_bf16.h>
#include <math.h>

// ---------------------------------------------------------------------------
// TernaryGRU for MI455X (gfx1250, wave32, WMMA + async global->LDS staging)
//   B=128, T=512, D=512, U=512, G=3U=1536
// ---------------------------------------------------------------------------

#define B_   128
#define T_   512
#define D_   512
#define U_   512
#define G_   1536           // 3*U
#define M_   (B_ * T_)      // 65536 rows of the big GEMM
#define WN_  (D_ * G_)      // 786432 elements per weight matrix
#define KSTEPS_ (D_ / 32)   // 16 WMMA K-steps
#define APAD_ 40            // padded LDS row stride (halves) to dodge bank conflicts

typedef __attribute__((ext_vector_type(16))) _Float16 v16h;
typedef __attribute__((ext_vector_type(8)))  float    v8f;

union Frag {            // 32 bytes: one WMMA 16-bit A/B fragment per lane
    v16h  h;
    uint4 q[2];
};

struct Half4 { _Float16 a, b, c, d; };   // 8-byte packed store

__device__ __forceinline__ float sigmoidf_(float x) {
    return 1.0f / (1.0f + __expf(-x));
}

// ----- CDNA5 async copy: 16B global -> LDS, tracked by ASYNCcnt -------------
__device__ __forceinline__ void async_b128(void* lds_ptr, const void* gptr) {
    unsigned lds_off = (unsigned)(size_t)lds_ptr;   // low 32 bits = LDS address
    asm volatile("global_load_async_to_lds_b128 %0, %1, off"
                 :: "v"(lds_off), "v"(gptr) : "memory");
}
__device__ __forceinline__ void wait_async0() {
    asm volatile("s_wait_asynccnt 0" ::: "memory");
}

// A fragment (16x32 f16): two 16B chunks 32B apart (K and K+16 halves)
__device__ __forceinline__ void loadA(Frag& f, const _Float16* p) {
    const uint4* q = reinterpret_cast<const uint4*>(p);
    f.q[0] = q[0];
    f.q[1] = q[2];
}
// B fragment (32x16 f16, g-major weights): two 16B chunks 32B apart (K and K+16 halves)
__device__ __forceinline__ void loadB(Frag& f, const _Float16* p) {
    const uint4* q = reinterpret_cast<const uint4*>(p);
    f.q[0] = q[0];
    f.q[1] = q[2];
}

// ---------------------------------------------------------------------------
// 1) alpha = mean(|w|) for both weight matrices. Deterministic tree reduce.
// ---------------------------------------------------------------------------
__global__ __launch_bounds__(256)
void absmean_kernel(const float* __restrict__ w0, const float* __restrict__ w1,
                    float* __restrict__ alphas) {
    const float* w = (blockIdx.x == 0) ? w0 : w1;
    __shared__ float sdata[256];
    float s = 0.0f;
    for (int i = threadIdx.x; i < WN_; i += 256) s += fabsf(w[i]);
    sdata[threadIdx.x] = s;
    __syncthreads();
    for (int off = 128; off > 0; off >>= 1) {
        if (threadIdx.x < off) sdata[threadIdx.x] += sdata[threadIdx.x + off];
        __syncthreads();
    }
    if (threadIdx.x == 0) alphas[blockIdx.x] = sdata[0] / (float)WN_;
}

// ---------------------------------------------------------------------------
// 2) Ternarize to fp16 ({-1,0,+1} exact) and TRANSPOSE to g-major so that
//    WMMA B-fragments are contiguous-K loads. alpha applied in GEMM epilogues.
// ---------------------------------------------------------------------------
__global__ __launch_bounds__(256)
void quantize_t_kernel(const float* __restrict__ w0, const float* __restrict__ w1,
                       const float* __restrict__ alphas,
                       _Float16* __restrict__ q0, _Float16* __restrict__ q1) {
    int i = blockIdx.x * 256 + threadIdx.x;
    if (i >= 2 * WN_) return;
    const float* w; _Float16* q; float a; int li;
    if (i < WN_) { w = w0; q = q0; a = alphas[0]; li = i; }
    else         { w = w1; q = q1; a = alphas[1]; li = i - WN_; }
    float v   = w[li];
    float thr = 0.7f * a;
    float t   = (fabsf(v) < thr) ? 0.0f : ((v > 0.0f) ? 1.0f : -1.0f);
    int k = li / G_;            // 0..511   (row of (D,3U))
    int g = li % G_;            // 0..1535  (col)
    q[(size_t)g * D_ + k] = (_Float16)t;   // transposed: [g][k]
}

// ---------------------------------------------------------------------------
// 3) x -> fp16 (vectorized)
// ---------------------------------------------------------------------------
__global__ __launch_bounds__(256)
void cvt_x_kernel(const float* __restrict__ x, _Float16* __restrict__ x16) {
    size_t i = ((size_t)blockIdx.x * 256 + threadIdx.x) * 4;
    float4 v = *reinterpret_cast<const float4*>(x + i);
    Half4 h; h.a = (_Float16)v.x; h.b = (_Float16)v.y;
             h.c = (_Float16)v.z; h.d = (_Float16)v.w;
    *reinterpret_cast<Half4*>(x16 + i) = h;
}

// ---------------------------------------------------------------------------
// 4) Big GEMM: xp[65536,1536] = alpha_k * (x16 @ wT) + bias0
//    Block = 256 threads = 8 waves (4 M x 2 N); block tile 64x128.
//    Per K-step (K=32): A slab 64x32 and B slab 128x32 are staged into
//    double-buffered LDS with global_load_async_to_lds_b128 (ASYNCcnt),
//    overlapping the previous step's WMMAs; fragments come from ds_load_b128.
// ---------------------------------------------------------------------------
__global__ __launch_bounds__(256)
void gemm_xp_kernel(const _Float16* __restrict__ x16,
                    const _Float16* __restrict__ wq,      // [g][k] g-major
                    const float* __restrict__ bias,       // bias[0][g]
                    const float* __restrict__ alphas,
                    float* __restrict__ xp) {
    __shared__ _Float16 sA[2][64  * APAD_];   // 2 x 5 KB
    __shared__ _Float16 sB[2][128 * APAD_];   // 2 x 10 KB

    const int tid  = threadIdx.x;
    const int wave = tid >> 5;
    const int lane = tid & 31;
    const int ln   = lane & 15;
    const int lh   = lane >> 4;
    const int wm   = wave & 3;                  // M sub-tile (0..3)
    const int wn   = wave >> 2;                 // N sub-tile (0..1)
    const int rowBase0 = blockIdx.x * 64;
    const int colBase0 = blockIdx.y * 128;

    // staging decomposition (per K-step slab):
    //   A: 64 rows x 32 halves = 256 x 16B  -> 1 async b128 per thread
    //   B: 128 cols x 32 halves = 512 x 16B -> 2 async b128 per thread
    const int ar  = tid >> 2;                   // A row   0..63
    const int as  = tid & 3;                    // A 16B segment 0..3
    const int bc  = tid >> 1;                   // B col   0..127
    const int bs0 = (tid & 1) * 2;              // B segments {0,1} or {2,3}
    const _Float16* agsrc = x16 + (size_t)(rowBase0 + ar) * D_ + as * 8;
    const _Float16* bgsrc = wq + (size_t)(colBase0 + bc) * D_;

    v8f acc[4] = {};

    // prologue: stage K-step 0 into buffer 0
    async_b128(&sA[0][ar * APAD_ + as * 8],        agsrc);
    async_b128(&sB[0][bc * APAD_ + bs0 * 8],       bgsrc + bs0 * 8);
    async_b128(&sB[0][bc * APAD_ + (bs0 + 1) * 8], bgsrc + (bs0 + 1) * 8);

#pragma unroll
    for (int ks = 0; ks < KSTEPS_; ++ks) {
        const int cur = ks & 1;
        const int nxt = cur ^ 1;

        wait_async0();       // own stages into cur complete
        __syncthreads();     // everyone's stages visible; nxt free to overwrite

        if (ks + 1 < KSTEPS_) {
            const int kn = (ks + 1) * 32;
            async_b128(&sA[nxt][ar * APAD_ + as * 8],        agsrc + kn);
            async_b128(&sB[nxt][bc * APAD_ + bs0 * 8],       bgsrc + kn + bs0 * 8);
            async_b128(&sB[nxt][bc * APAD_ + (bs0 + 1) * 8], bgsrc + kn + (bs0 + 1) * 8);
        }

        Frag fa;
        loadA(fa, &sA[cur][(wm * 16 + ln) * APAD_ + lh * 8]);
#pragma unroll
        for (int t = 0; t < 4; ++t) {
            Frag fb;
            loadB(fb, &sB[cur][(wn * 64 + t * 16 + ln) * APAD_ + lh * 8]);
            acc[t] = __builtin_amdgcn_wmma_f32_16x16x32_f16(
                false, fa.h, false, fb.h, (short)0, acc[t], false, false);
        }
    }

    const float alpha = alphas[0];
#pragma unroll
    for (int t = 0; t < 4; ++t) {
        const int col = colBase0 + wn * 64 + t * 16 + ln;
        const float b0 = bias[col];
#pragma unroll
        for (int v = 0; v < 8; ++v) {
            const int row = rowBase0 + wm * 16 + v + lh * 8;
            // xp is 402 MB streamed once -> non-temporal, keep L2 for weights
            __builtin_nontemporal_store(alpha * acc[t][v] + b0,
                                        &xp[(size_t)row * G_ + col]);
        }
    }
}

// ---------------------------------------------------------------------------
// 5) zero-init h state (buffer 0)
// ---------------------------------------------------------------------------
__global__ __launch_bounds__(256)
void init_h_kernel(float* __restrict__ h32, _Float16* __restrict__ h16) {
    int i = blockIdx.x * 256 + threadIdx.x;
    if (i < B_ * U_) { h32[i] = 0.0f; h16[i] = (_Float16)0.0f; }
}

// ---------------------------------------------------------------------------
// 6) One GRU step. 32 blocks x 8 waves = 256 waves; wave (mt,j) owns batch
//    rows [16mt,16mt+16) x U cols [16j,16j+16). Three WMMA accumulators
//    (z/r/h gate columns) share one A fragment; gates fused in registers.
//    Double-buffered h (in -> out) so there is no intra-step race, and
//    double-buffered fragments so loads overlap WMMA (weights are L2-hot).
// ---------------------------------------------------------------------------
__global__ __launch_bounds__(256)
void gru_step_kernel(const _Float16* __restrict__ h16_in,
                     const float*    __restrict__ h32_in,
                     _Float16*       __restrict__ h16_out,
                     float*          __restrict__ h32_out,
                     const _Float16* __restrict__ rq,    // [g][k] g-major
                     const float*    __restrict__ xp,
                     const float*    __restrict__ bias,  // [2][1536]
                     const float*    __restrict__ alphas,
                     int tstep) {
    const int wave = threadIdx.x >> 5;
    const int lane = threadIdx.x & 31;
    const int ln   = lane & 15;
    const int lh   = lane >> 4;
    const int task = blockIdx.x * 8 + wave;   // 0..255
    const int mt   = task >> 5;               // 0..7  batch tile
    const int j    = task & 31;               // 0..31 U tile
    const int rowBase = mt * 16;
    const int ucol    = j * 16 + ln;          // 0..511

    v8f az = {}, ar = {}, ah = {};
    const _Float16* aptr = h16_in + (size_t)(rowBase + ln) * U_ + lh * 8;
    const _Float16* bz = rq + (size_t)(0 * U_ + ucol) * D_ + lh * 8;
    const _Float16* br = rq + (size_t)(1 * U_ + ucol) * D_ + lh * 8;
    const _Float16* bh = rq + (size_t)(2 * U_ + ucol) * D_ + lh * 8;

    Frag fa[2], fz[2], fr[2], fh[2];
    loadA(fa[0], aptr);
    loadB(fz[0], bz);
    loadB(fr[0], br);
    loadB(fh[0], bh);

#pragma unroll
    for (int ks = 0; ks < KSTEPS_; ++ks) {
        const int cur = ks & 1;
        const int nxt = cur ^ 1;
        if (ks + 1 < KSTEPS_) {
            const int kn = (ks + 1) * 32;
            loadA(fa[nxt], aptr + kn);
            loadB(fz[nxt], bz + kn);
            loadB(fr[nxt], br + kn);
            loadB(fh[nxt], bh + kn);
        }
        az = __builtin_amdgcn_wmma_f32_16x16x32_f16(
            false, fa[cur].h, false, fz[cur].h, (short)0, az, false, false);
        ar = __builtin_amdgcn_wmma_f32_16x16x32_f16(
            false, fa[cur].h, false, fr[cur].h, (short)0, ar, false, false);
        ah = __builtin_amdgcn_wmma_f32_16x16x32_f16(
            false, fa[cur].h, false, fh[cur].h, (short)0, ah, false, false);
    }

    const float alpha = alphas[1];
    const float rbz = bias[G_ + ucol];
    const float rbr = bias[G_ + U_ + ucol];
    const float rbh = bias[G_ + 2 * U_ + ucol];

#pragma unroll
    for (int v = 0; v < 8; ++v) {
        const int b = rowBase + v + lh * 8;
        const size_t xrow = ((size_t)b * T_ + tstep) * G_;
        // xp rows are read exactly once across the whole scan -> non-temporal
        const float xz = __builtin_nontemporal_load(&xp[xrow + ucol]);
        const float xr = __builtin_nontemporal_load(&xp[xrow + U_ + ucol]);
        const float xh = __builtin_nontemporal_load(&xp[xrow + 2 * U_ + ucol]);
        const float iz = alpha * az[v] + rbz;
        const float ir = alpha * ar[v] + rbr;
        const float ih = alpha * ah[v] + rbh;
        const float z  = sigmoidf_(xz + iz);
        const float r  = sigmoidf_(xr + ir);
        const float hh = tanhf(xh + r * ih);
        const size_t hidx = (size_t)b * U_ + ucol;
        const float hold = h32_in[hidx];
        const float hnew = z * hold + (1.0f - z) * hh;
        h32_out[hidx] = hnew;
        h16_out[hidx] = (_Float16)hnew;
    }
}

// ---------------------------------------------------------------------------
// Workspace layout (bytes, 256-aligned)
// ---------------------------------------------------------------------------
static const size_t OFF_ALPHA = 0;                                   // 256 B
static const size_t OFF_WQ    = 256;                                 // 1.5 MB
static const size_t OFF_RQ    = OFF_WQ  + (size_t)WN_ * 2;
static const size_t OFF_X16   = OFF_RQ  + (size_t)WN_ * 2;           // 64 MB
static const size_t OFF_H32A  = OFF_X16 + (size_t)M_ * D_ * 2;
static const size_t OFF_H32B  = OFF_H32A + (size_t)B_ * U_ * 4;
static const size_t OFF_H16A  = OFF_H32B + (size_t)B_ * U_ * 4;
static const size_t OFF_H16B  = OFF_H16A + (size_t)B_ * U_ * 2;
static const size_t OFF_XP    = OFF_H16B + (size_t)B_ * U_ * 2;      // 384 MB

extern "C" void kernel_launch(void* const* d_in, const int* in_sizes, int n_in,
                              void* d_out, int out_size, void* d_ws, size_t ws_size,
                              hipStream_t stream) {
    const float* x    = (const float*)d_in[0];
    const float* ker  = (const float*)d_in[1];
    const float* rker = (const float*)d_in[2];
    const float* bias = (const float*)d_in[3];

    char* ws = (char*)d_ws;
    float*    alphas = (float*)   (ws + OFF_ALPHA);
    _Float16* wq     = (_Float16*)(ws + OFF_WQ);
    _Float16* rq     = (_Float16*)(ws + OFF_RQ);
    _Float16* x16    = (_Float16*)(ws + OFF_X16);
    float*    h32a   = (float*)   (ws + OFF_H32A);
    float*    h32b   = (float*)   (ws + OFF_H32B);
    _Float16* h16a   = (_Float16*)(ws + OFF_H16A);
    _Float16* h16b   = (_Float16*)(ws + OFF_H16B);
    float*    xp     = (float*)   (ws + OFF_XP);

    // Phase 0: ternary stats + quantize (transposed) + x->fp16
    absmean_kernel<<<2, 256, 0, stream>>>(ker, rker, alphas);
    quantize_t_kernel<<<(2 * WN_) / 256, 256, 0, stream>>>(ker, rker, alphas, wq, rq);
    cvt_x_kernel<<<(M_ * D_ / 4) / 256, 256, 0, stream>>>(x, x16);

    // Phase 1: xp = alpha_k * (x16 @ wT) + bias0   (WMMA + async LDS staging)
    dim3 g1(M_ / 64, G_ / 128);
    gemm_xp_kernel<<<g1, 256, 0, stream>>>(x16, wq, bias, alphas, xp);

    // Phase 2: GRU scan, double-buffered state
    init_h_kernel<<<(B_ * U_ + 255) / 256, 256, 0, stream>>>(h32a, h16a);
    for (int t = 0; t < T_; ++t) {
        const bool even = (t & 1) == 0;
        const _Float16* hin16 = even ? h16a : h16b;
        const float*    hin32 = even ? h32a : h32b;
        _Float16*       ho16  = even ? h16b : h16a;
        float*          ho32  = even ? h32b : h32a;
        gru_step_kernel<<<32, 256, 0, stream>>>(hin16, hin32, ho16, ho32,
                                                rq, xp, bias, alphas, t);
    }

    // T=512 steps: final state lands back in buffer A
    hipMemcpyAsync(d_out, h32a, (size_t)B_ * U_ * sizeof(float),
                   hipMemcpyDeviceToDevice, stream);
}
